// ONNX_TRT_39333310496772
// MI455X (gfx1250) — matrix operand order repaired and checked by reading, not compiled, not run on hardware
//
#include <hip/hip_runtime.h>
#include <hip/hip_bf16.h>
#include <cstdint>
#include <cstddef>

#define NCLS 80
#define MAX_OBJn 100
#define BATCH 4
#define NA 25200
#define NMc 32
#define PWd 160
#define HWp (PWd * PWd)          // 25600
#define CCH (5 + NCLS + NMc)     // 117
#define SELTOT (BATCH * MAX_OBJn) // 400

// d_out layout (floats, tuple concatenated in return order)
#define OFF_BOXES   4
#define OFF_SCORES  (OFF_BOXES + SELTOT * 4)
#define OFF_CLASSES (OFF_SCORES + SELTOT)
#define OFF_MASKS   (OFF_CLASSES + SELTOT)

typedef float    v2f  __attribute__((ext_vector_type(2)));
typedef float    v8f  __attribute__((ext_vector_type(8)));
typedef _Float16 v16h __attribute__((ext_vector_type(16)));
typedef unsigned int u32x4 __attribute__((ext_vector_type(4)));
typedef int          i32x4 __attribute__((ext_vector_type(4)));
typedef int          i32x8 __attribute__((ext_vector_type(8)));

#if defined(__has_builtin)
#  if __has_builtin(__builtin_amdgcn_wmma_f32_16x16x4_f32)
#    define HAVE_WMMA_F32X4 1
#  endif
#  if __has_builtin(__builtin_amdgcn_tensor_load_to_lds) && __has_builtin(__builtin_amdgcn_s_wait_tensorcnt)
#    define USE_TDM 1
#  endif
#endif
#ifndef HAVE_WMMA_F32X4
#  define HAVE_WMMA_F32X4 0
#endif
#ifndef USE_TDM
#  define USE_TDM 0
#endif

// ---------------------------------------------------------------------------
// K1: gather the 400 selected rows of x0; box transform, max/argmax, mask coefs
// ---------------------------------------------------------------------------
__global__ void k1_gather_rows(const float* __restrict__ x0,
                               const int* __restrict__ sel,
                               const float* __restrict__ cm,
                               float* __restrict__ sel_boxes,
                               float* __restrict__ sel_scores,
                               float* __restrict__ sel_cat,
                               float* __restrict__ sel_mask) {
  int n = blockIdx.x * blockDim.x + threadIdx.x;
  if (n >= SELTOT) return;
  int X = sel[3 * n];
  int Y = sel[3 * n + 2];
  const float* row = x0 + ((size_t)X * NA + (size_t)Y) * CCH;
  float b0 = row[0], b1 = row[1], b2 = row[2], b3 = row[3];
#pragma unroll
  for (int j = 0; j < 4; ++j)
    sel_boxes[n * 4 + j] = b0 * cm[j] + b1 * cm[4 + j] + b2 * cm[8 + j] + b3 * cm[12 + j];
  float conf = row[4];
  float best = -3.402823466e+38f;
  int bc = 0;
  for (int c = 0; c < NCLS; ++c) {
    float s = row[5 + c] * conf;
    if (s > best) { best = s; bc = c; }   // first-max == jnp.argmax
  }
  sel_scores[n] = best;
  sel_cat[n] = (float)bc;
#pragma unroll
  for (int d = 0; d < NMc; ++d) sel_mask[n * NMc + d] = row[5 + NCLS + d];
}

// ---------------------------------------------------------------------------
// K2: sequential planning (num_object, num_det, idxs = top_k *values*, inv_slot)
// ---------------------------------------------------------------------------
__global__ void k2_plan(const int* __restrict__ sel,
                        int* __restrict__ inv_slot,
                        int* __restrict__ idxs,
                        int* __restrict__ num_det,
                        int* __restrict__ num_object_p,
                        float* __restrict__ out_numdet) {
  if (threadIdx.x != 0 || blockIdx.x != 0) return;
  int sums[SELTOT];
  for (int n = 0; n < SELTOT; ++n)
    sums[n] = sel[3 * n] + sel[3 * n + 1] + sel[3 * n + 2];
  int ln1 = -1, l2 = -1;
  for (int n = 0; n < SELTOT; ++n) if (sums[n] > 0) ln1 = n;
  for (int i = 0; i < SELTOT - 1; ++i) if (sums[i + 1] - sums[i] != 0) l2 = i;
  long lagsum = (long)sums[SELTOT - 1] - (long)sums[0];
  int no1 = (ln1 < 0 ? 0 : ln1) + 1;
  int no2 = (l2 < 0 ? 0 : l2) + 2;
  int numo = (lagsum != 0) ? (no1 < no2 ? no1 : no2) : 0;
  *num_object_p = numo;
  for (int n = 0; n < SELTOT; ++n) inv_slot[n] = -1;
  for (int b = 0; b < BATCH; ++b) {
    int j = 0, cnt = 0;
    for (int n = SELTOT - 1; n >= 0; --n) {
      if (n < numo && sel[3 * n] == b) {
        ++cnt;
        if (j < MAX_OBJn) { idxs[b * MAX_OBJn + j] = n; inv_slot[n] = b * MAX_OBJn + j; ++j; }
      }
    }
    for (; j < MAX_OBJn; ++j) idxs[b * MAX_OBJn + j] = 0;  // top_k zero padding
    num_det[b] = cnt;
    out_numdet[b] = (float)cnt;  // num_det output (int32 value stored as float)
  }
}

// ---------------------------------------------------------------------------
// K3: gather det_boxes / det_scores / det_classes through idxs
// ---------------------------------------------------------------------------
__global__ void k3_gather_out(const int* __restrict__ idxs,
                              const float* __restrict__ sel_boxes,
                              const float* __restrict__ sel_scores,
                              const float* __restrict__ sel_cat,
                              float* __restrict__ out) {
  int s = blockIdx.x * blockDim.x + threadIdx.x;
  if (s >= SELTOT) return;
  int n = idxs[s];
#pragma unroll
  for (int k = 0; k < 4; ++k) out[OFF_BOXES + s * 4 + k] = sel_boxes[n * 4 + k];
  out[OFF_SCORES + s] = sel_scores[n];
  out[OFF_CLASSES + s] = sel_cat[n];
}

// ---------------------------------------------------------------------------
// K4: mask GEMM via WMMA. Block = 8 waves, 16 selection-rows x 128 cols.
//     B tile (32x128 f32 = 16KB) staged to LDS via the Tensor Data Mover.
// ---------------------------------------------------------------------------
__global__ __launch_bounds__(256) void k4_mask_wmma(
    const float* __restrict__ proto,
    const int* __restrict__ sel,
    const float* __restrict__ sel_mask,
    const float* __restrict__ sel_boxes,
    const int* __restrict__ inv_slot,
    const int* __restrict__ num_object_p,
    float* __restrict__ out_masks,
    float* __restrict__ mask0) {
  const int b = blockIdx.z;
  const int rowbase = blockIdx.y * 16;
  const int colbase = blockIdx.x * 128;

  __shared__ float s_db[16][4];
  __shared__ int   s_out[16];
  __shared__ int   s_flag;
  __shared__ int   s_x0;
#if USE_TDM
  __shared__ float ldsB[NMc * 128];   // 16 KB proto tile
#endif

  const int t = threadIdx.x;
  if (t == 0) { s_flag = 0; s_x0 = sel[0]; }
  __syncthreads();
  if (t < 16) {
    int m = rowbase + t;
    int Xm = sel[3 * m];
    int numo = *num_object_p;
    int o = inv_slot[m];
    int oo = (Xm == b && m < numo && o >= 0) ? o : -1;
    s_out[t] = oo;
#pragma unroll
    for (int k = 0; k < 4; ++k) s_db[t][k] = sel_boxes[m * 4 + k] * 0.25f;
    if (oo >= 0 || (m == 0 && Xm == b)) atomicOr(&s_flag, 1);
  }
  __syncthreads();
  if (!s_flag) return;   // uniform early-out: nothing in this tile belongs to batch b

  const int lane = t & 31;
  const int wv   = t >> 5;
  const int kh   = (lane >> 4) * 2;          // f32 A/B layout: upper half-wave holds K+2
  const int lcol = wv * 16 + (lane & 15);    // column within the 128-wide tile

#if USE_TDM
  if (wv == 0) {
    // Build Tensor DMA Descriptor (D#): 2D tile, 4-byte elements.
    unsigned lds_off = (unsigned)(size_t)(&ldsB[0]);   // low 32 bits of generic = LDS addr
    unsigned long long ga =
        (unsigned long long)(const void*)(proto + ((size_t)b * NMc) * HWp + colbase);
    u32x4 g0;
    g0[0] = 1u;                                         // count=1, user mode
    g0[1] = lds_off;                                    // lds_addr
    g0[2] = (unsigned)(ga & 0xFFFFFFFFu);               // global_addr[31:0]
    g0[3] = (unsigned)((ga >> 32) & 0x01FFFFFFu) | (2u << 30);  // addr[56:32] | type=2
    int td0 = HWp - colbase;                            // remaining tensor width
    i32x8 g1;
    g1[0] = (int)(2u << 16);                            // workgroup_mask=0, data_size=4B
    g1[1] = (int)((td0 & 0xFFFF) << 16);                // tensor_dim0 lo16
    g1[2] = (int)(((td0 >> 16) & 0xFFFF) | (NMc << 16));// tensor_dim0 hi16 | tensor_dim1 lo16
    g1[3] = (int)(128 << 16);                           // tensor_dim1 hi16=0 | tile_dim0=128
    g1[4] = NMc;                                        // tile_dim1=32 | tile_dim2=0
    g1[5] = HWp;                                        // tensor_dim0_stride lo32
    g1[6] = 0;                                          // stride hi16 | dim1_stride lo16
    g1[7] = 0;
    i32x4 z4 = {};
    i32x8 z8 = {};
    __builtin_amdgcn_tensor_load_to_lds(g0, g1, z4, z4, z8, 0);  // 6-arg toolchain form
    __builtin_amdgcn_s_wait_tensorcnt(0);
  }
  __syncthreads();
#define B_ELEM(k) ldsB[(k) * 128 + lcol]
#else
  const size_t protoBase = ((size_t)b * NMc) * HWp + (size_t)(colbase + lcol);
#define B_ELEM(k) proto[protoBase + (size_t)(k) * HWp]
#endif

  const int mrow = rowbase + (lane & 15);
  const float* __restrict__ arow = sel_mask + (size_t)mrow * NMc;
  v8f acc = {};

#if HAVE_WMMA_F32X4
  // fp32-exact: 8 x V_WMMA_F32_16X16X4_F32, K = 32
  float areg[16];
#pragma unroll
  for (int s = 0; s < 8; ++s) {
    areg[2 * s]     = arow[s * 4 + kh];
    areg[2 * s + 1] = arow[s * 4 + kh + 1];
  }
#pragma unroll
  for (int s = 0; s < 8; ++s) {
    v2f a; a.x = areg[2 * s]; a.y = areg[2 * s + 1];
    v2f bf; bf.x = B_ELEM(s * 4 + kh); bf.y = B_ELEM(s * 4 + kh + 1);
    acc = __builtin_amdgcn_wmma_f32_16x16x4_f32(false, a, false, bf, (short)0, acc,
                                                false, false);
  }
#else
  // fallback: single V_WMMA_F32_16X16X32_F16 (K = 32)
  const int ka = (lane >> 4) * 16;  // A halves: lower lanes K0..15, upper K16..31
  v16h ah, bh;
#pragma unroll
  for (int e = 0; e < 16; ++e) {
    ah[e]     = (_Float16)arow[ka + e];
  }
  const int kb = (lane >> 4) * 16;  // B halves: lower lanes K0..15, upper K16..31
#pragma unroll
  for (int e = 0; e < 16; ++e) bh[e] = (_Float16)B_ELEM(kb + e);
  acc = __builtin_amdgcn_wmma_f32_16x16x32_f16(false, ah, false, bh, (short)0, acc,
                                               false, false);
#endif
#undef B_ELEM

  // Epilogue: sigmoid, box crop, scatter to output slots (+ mask0 capture)
  const int col = colbase + lcol;
  const float px = (float)(col % PWd);   // width index vs x1/x2
  const float py = (float)(col / PWd);   // height index vs y1/y2
  const int x0b = s_x0;
#pragma unroll
  for (int r = 0; r < 8; ++r) {
    int ml = r + ((lane >> 4) * 8);      // local row 0..15 (C/D layout)
    int oo = s_out[ml];
    int m = rowbase + ml;
    bool is0 = (m == 0) && (x0b == b);
    if (oo < 0 && !is0) continue;
    float v = 1.0f / (1.0f + __expf(-acc[r]));
    float x1 = s_db[ml][0], y1 = s_db[ml][1], x2 = s_db[ml][2], y2 = s_db[ml][3];
    float crop = (px >= x1 && px < x2 && py >= y1 && py < y2) ? 1.0f : 0.0f;
    v *= crop;
    if (oo >= 0) out_masks[(size_t)oo * HWp + col] = v;
    if (is0) mask0[col] = v;
  }
}

// ---------------------------------------------------------------------------
// K5: fill padding slots (j >= num_det[b]) with masks[idx 0]
// ---------------------------------------------------------------------------
__global__ void k5_pad_fill(const int* __restrict__ num_det,
                            const float* __restrict__ mask0,
                            float* __restrict__ out_masks) {
  int slot = blockIdx.x;            // 0..399
  int b = slot / MAX_OBJn, j = slot % MAX_OBJn;
  if (j < num_det[b]) return;
  float* dst = out_masks + (size_t)slot * HWp;
  for (int c = threadIdx.x; c < HWp; c += blockDim.x) dst[c] = mask0[c];
}

// ---------------------------------------------------------------------------
extern "C" void kernel_launch(void* const* d_in, const int* in_sizes, int n_in,
                              void* d_out, int out_size, void* d_ws, size_t ws_size,
                              hipStream_t stream) {
  const float* x0    = (const float*)d_in[0];
  const float* proto = (const float*)d_in[1];
  const int*   sel   = (const int*)d_in[2];
  const float* cm    = (const float*)d_in[3];
  float* out = (float*)d_out;

  // workspace layout (bytes)
  char* ws = (char*)d_ws;
  float* sel_boxes  = (float*)(ws + 0);        // 400*4 f32
  float* sel_scores = (float*)(ws + 6400);     // 400 f32
  float* sel_cat    = (float*)(ws + 8000);     // 400 f32
  float* sel_mask   = (float*)(ws + 9600);     // 400*32 f32 -> ends 60800
  int*   inv_slot   = (int*)(ws + 60800);      // 400 i32
  int*   idxs       = (int*)(ws + 62400);      // 400 i32
  int*   num_det    = (int*)(ws + 64000);      // 4 i32
  int*   num_obj    = (int*)(ws + 64016);      // 1 i32
  float* mask0      = (float*)(ws + 64064);    // 25600 f32 -> total ~166 KB

  k1_gather_rows<<<dim3((SELTOT + 255) / 256), dim3(256), 0, stream>>>(
      x0, sel, cm, sel_boxes, sel_scores, sel_cat, sel_mask);

  k2_plan<<<dim3(1), dim3(32), 0, stream>>>(
      sel, inv_slot, idxs, num_det, num_obj, out /* out[0..3] = num_det */);

  k3_gather_out<<<dim3((SELTOT + 255) / 256), dim3(256), 0, stream>>>(
      idxs, sel_boxes, sel_scores, sel_cat, out);

  k4_mask_wmma<<<dim3(HWp / 128, SELTOT / 16, BATCH), dim3(256), 0, stream>>>(
      proto, sel, sel_mask, sel_boxes, inv_slot, num_obj,
      out + OFF_MASKS, mask0);

  k5_pad_fill<<<dim3(SELTOT), dim3(256), 0, stream>>>(
      num_det, mask0, out + OFF_MASKS);
}
